// EngineA_Small_52450140619166
// MI455X (gfx1250) — matrix operand order repaired and checked by reading, not compiled, not run on hardware
//
#include <hip/hip_runtime.h>
#include <hip/hip_bf16.h>
#include <math.h>

// ---------------------------------------------------------------------------
// MoE (top-2 of 6 experts) for gfx1250 / MI455X.
//   K0: zero d_out + routing counters
//   K1: gating + per-expert token lists
//   (ws large)  Kc: convert x->f16, W1->[e][h][i] f16, W2->[e][o][h] f16
//               K2a: fused expert kernel, staging via global_load_async_to_lds
//   (ws small)  K2b: fused expert kernel, inline fp32->f16 staging (fallback)
// f16 WMMA 16x16x32 with f32 accumulation throughout.
// ---------------------------------------------------------------------------

typedef _Float16 h16;
typedef __attribute__((ext_vector_type(16))) _Float16 v16h;
typedef __attribute__((ext_vector_type(8)))  _Float16 v8h;
typedef __attribute__((ext_vector_type(4)))  _Float16 v4h;
typedef __attribute__((ext_vector_type(2)))  __fp16   pkh2;   // cvt_pkrtz result
typedef __attribute__((ext_vector_type(8)))  float    v8f;
typedef __attribute__((ext_vector_type(4)))  float    v4f;

#define NEXP     6
#define IN_DIM   1024
#define HID_DIM  4096
#define OUT_DIM  1024
#define NTOK     8192
#define MTILE    32

// LDS strides (halfs): every b128 fragment access 16B-aligned, strides chosen
// (XS/HS 4, BS 20 dwords mod 64 banks) for conflict-free half-wave access.
#define XS_STRIDE  1032
#define HS_STRIDE  136
#define BS_STRIDE  40

#define XS_HALFS   (32 * XS_STRIDE)
#define HS_HALFS   (32 * HS_STRIDE)
#define BS_HALFS   (2 * 128 * BS_STRIDE)
#define W2S_HALFS  (2 * 1024 * BS_STRIDE)
#define SMEM_BYTES ((XS_HALFS + HS_HALFS + BS_HALFS + W2S_HALFS) * 2 + 32 * 4 + 32 * 4)

// ---------------- CDNA5 async global->LDS (CDNA5 ISA) -------------
__device__ __forceinline__ void async_g2l_b128(unsigned lds, unsigned long long g) {
    asm volatile("global_load_async_to_lds_b128 %0, %1, off"
                 :: "v"(lds), "v"(g) : "memory");
}
#define WAIT_ASYNC(n) asm volatile("s_wait_asynccnt " #n ::: "memory")

// ---------------- fragment helpers (contiguous-K order, not the ISA 7.12.2 table order) -----------------
// A (16x32 f16): lane l = row m (l%16); 16 contiguous K at K0 = 16*(l/16)
__device__ __forceinline__ v16h ld_a(const h16* p) {
    union { v16h v; v8h h[2]; } u;
    u.h[0] = *(const v8h*)(p);
    u.h[1] = *(const v8h*)(p + 8);
    return u.v;
}
// B (32x16 f16): lane l = col n (l%16); 16 contiguous K at K0 = 16*(l/16)
__device__ __forceinline__ v16h ld_b(const h16* p) {
    union { v16h v; v8h h[2]; } u;
    u.h[0] = *(const v8h*)(p);
    u.h[1] = *(const v8h*)(p + 8);
    return u.v;
}
__device__ __forceinline__ v8f wmma16x16x32(v16h a, v16h b, v8f c) {
    return __builtin_amdgcn_wmma_f32_16x16x32_f16(false, a, false, b,
                                                  (short)0, c, false, false);
}
__device__ __forceinline__ v8h relu_pack(v8f c, const float* bb) {
    union { v8h v; pkh2 h[4]; } u;
#pragma unroll
    for (int i = 0; i < 4; ++i)
        u.h[i] = __builtin_amdgcn_cvt_pkrtz(fmaxf(c[2 * i] + bb[2 * i], 0.f),
                                            fmaxf(c[2 * i + 1] + bb[2 * i + 1], 0.f));
    return u.v;
}

// ---------------- K0: zero output + counters -------------------------------
__global__ void k_zero(float* __restrict__ out, int* __restrict__ counts) {
    int i = blockIdx.x * blockDim.x + threadIdx.x;
    v4f z = {0.f, 0.f, 0.f, 0.f};
    ((v4f*)out)[i] = z;
    if (blockIdx.x == 0 && threadIdx.x < NEXP) counts[threadIdx.x] = 0;
}

// ---------------- K1: gating + routing (one wave32 per token) ---------------
__global__ __launch_bounds__(256) void k_gate(
        const float* __restrict__ x, const float* __restrict__ Wg,
        const float* __restrict__ bg, int* __restrict__ counts,
        int* __restrict__ tlist, float* __restrict__ wlist) {
    const int lane = threadIdx.x & 31;
    const int t    = blockIdx.x * 8 + (threadIdx.x >> 5);
    if (t >= NTOK) return;

    float acc[NEXP] = {0.f, 0.f, 0.f, 0.f, 0.f, 0.f};
    const float* xr = x + (size_t)t * IN_DIM;
    for (int i = lane; i < IN_DIM; i += 32) {
        float xv = xr[i];
        const float* wr = Wg + i * NEXP;
#pragma unroll
        for (int e = 0; e < NEXP; ++e) acc[e] += xv * wr[e];
    }
#pragma unroll
    for (int e = 0; e < NEXP; ++e)
        for (int off = 16; off > 0; off >>= 1)
            acc[e] += __shfl_xor(acc[e], off, 32);

    if (lane == 0) {
        float s[NEXP], mx = -1e30f;
#pragma unroll
        for (int e = 0; e < NEXP; ++e) { s[e] = acc[e] + bg[e]; mx = fmaxf(mx, s[e]); }
        int i1 = 0;
        for (int e = 1; e < NEXP; ++e) if (s[e] > s[i1]) i1 = e;
        int i2 = -1;
        for (int e = 0; e < NEXP; ++e) {
            if (e == i1) continue;
            if (i2 < 0 || s[e] > s[i2]) i2 = e;
        }
        float den = 0.f, ex[NEXP];
#pragma unroll
        for (int e = 0; e < NEXP; ++e) { ex[e] = expf(s[e] - mx); den += ex[e]; }
        float w1 = ex[i1] / den, w2 = ex[i2] / den;
        float rs = w1 + w2 + 1e-8f;
        w1 /= rs; w2 /= rs;
        int p1 = atomicAdd(&counts[i1], 1);
        tlist[i1 * NTOK + p1] = t; wlist[i1 * NTOK + p1] = w1;
        int p2 = atomicAdd(&counts[i2], 1);
        tlist[i2 * NTOK + p2] = t; wlist[i2 * NTOK + p2] = w2;
    }
}

// ---------------- Kc: conversion pass ---------------------------------------
__global__ void k_cvt_x(const float* __restrict__ x, h16* __restrict__ xh) {
    int i = blockIdx.x * blockDim.x + threadIdx.x;       // float4 index
    v4f v = ((const v4f*)x)[i];
    union { v4h v4; pkh2 h[2]; } u;
    u.h[0] = __builtin_amdgcn_cvt_pkrtz(v[0], v[1]);
    u.h[1] = __builtin_amdgcn_cvt_pkrtz(v[2], v[3]);
    ((v4h*)xh)[i] = u.v4;
}

// [e][R][C] fp32 -> [e][C][R] f16, 32x32 LDS tiles
__global__ __launch_bounds__(256) void k_cvt_t(const float* __restrict__ src,
                                               h16* __restrict__ dst,
                                               int R, int C) {
    __shared__ float tile[32][33];
    const int tilesC   = C >> 5;
    const int tilesPer = (R >> 5) * tilesC;
    const int b  = blockIdx.x;
    const int e  = b / tilesPer;
    const int t  = b - e * tilesPer;
    const int rt = t / tilesC, ct = t - rt * tilesC;
    const float* sb = src + (size_t)e * R * C;
    h16*         db = dst + (size_t)e * R * C;
    const int r  = threadIdx.x >> 3;
    const int c4 = (threadIdx.x & 7) * 4;
    v4f v = *(const v4f*)(sb + (size_t)(rt * 32 + r) * C + ct * 32 + c4);
    tile[r][c4 + 0] = v[0]; tile[r][c4 + 1] = v[1];
    tile[r][c4 + 2] = v[2]; tile[r][c4 + 3] = v[3];
    __syncthreads();
    union { v4h v4; pkh2 h[2]; } u;
    u.h[0] = __builtin_amdgcn_cvt_pkrtz(tile[c4 + 0][r], tile[c4 + 1][r]);
    u.h[1] = __builtin_amdgcn_cvt_pkrtz(tile[c4 + 2][r], tile[c4 + 3][r]);
    *(v4h*)(db + (size_t)(ct * 32 + r) * R + rt * 32 + c4) = u.v4;
}

// ---------------- async staging (f16 ws, K-contiguous layouts) --------------
// W1 chunk: w1e[h][i], rows h0..h0+127, K = k*32..+32 -> Bs[c][kk], 2 asyncs/thr
__device__ __forceinline__ void stage_w1_async(h16* dst, const h16* w1e,
                                               int h0, int k, int tid) {
#pragma unroll
    for (int j = 0; j < 2; ++j) {
        int q = tid * 2 + j;                 // 0..511 16B-chunks
        int col = q >> 2, qq = q & 3;
        unsigned long long g = (unsigned long long)(uintptr_t)
            (w1e + (size_t)(h0 + col) * IN_DIM + k * 32 + qq * 8);
        unsigned l = (unsigned)(uintptr_t)(dst + col * BS_STRIDE + qq * 8);
        async_g2l_b128(l, g);
    }
}
// W2 chunk: w2e[o][h], o=0..1023, K = h0+k2*32..+32 -> W2s[o][kk], 16 asyncs/thr
__device__ __forceinline__ void stage_w2_async(h16* dst, const h16* w2e,
                                               int h0, int k2, int tid) {
#pragma unroll
    for (int j = 0; j < 16; ++j) {
        int q = tid + 256 * j;               // 0..4095 16B-chunks
        int col = q >> 2, qq = q & 3;
        unsigned long long g = (unsigned long long)(uintptr_t)
            (w2e + (size_t)col * HID_DIM + h0 + k2 * 32 + qq * 8);
        unsigned l = (unsigned)(uintptr_t)(dst + col * BS_STRIDE + qq * 8);
        async_g2l_b128(l, g);
    }
}

// ---------------- K2a: fused expert kernel (async path) ---------------------
__global__ __launch_bounds__(256) void k_moe_async(
        const h16* __restrict__ xh, const h16* __restrict__ w1t,
        const h16* __restrict__ w2t,
        const float* __restrict__ b1, const float* __restrict__ b2,
        const int* __restrict__ counts, const int* __restrict__ tlist,
        const float* __restrict__ wlist, float* __restrict__ out) {
    const int e     = blockIdx.x >> 8;
    const int tile  = blockIdx.x & 255;
    const int nrows = counts[e];
    const int row0  = tile * MTILE;
    if (row0 >= nrows) return;

    extern __shared__ h16 smem[];
    h16* Xs  = smem;                       // [32][XS_STRIDE]  f16 tokens
    h16* Hs  = Xs + XS_HALFS;              // [32][HS_STRIDE]  hidden chunk
    h16* Bs  = Hs + HS_HALFS;              // [2][128][BS_STRIDE] W1 chunk (col-major)
    h16* W2s = Bs + BS_HALFS;              // [2][1024][BS_STRIDE] W2 chunk (col-major)
    int*   tok = (int*)(W2s + W2S_HALFS);
    float* wt  = (float*)(tok + 32);

    const int tid  = threadIdx.x;
    const int wv   = tid >> 5;
    const int lane = tid & 31;
    const int ln16 = lane & 15;
    const int lh   = lane >> 4;

    if (tid < 32) {
        int r = row0 + tid;
        int tk = 0; float w = 0.f;
        if (r < nrows) { tk = tlist[e * NTOK + r]; w = wlist[e * NTOK + r]; }
        tok[tid] = tk; wt[tid] = w;
    }
    __syncthreads();

    // gather X tile rows via async copy (already f16)
    {
        const int m = tid >> 3, g = tid & 7;
        unsigned long long grow = (unsigned long long)(uintptr_t)
            (xh + (size_t)tok[m] * IN_DIM);
        unsigned lrow = (unsigned)(uintptr_t)(Xs + m * XS_STRIDE);
#pragma unroll
        for (int j = 0; j < 16; ++j) {
            int c16 = g + 8 * j;
            async_g2l_b128(lrow + c16 * 16, grow + c16 * 16);
        }
    }

    const h16* w1e = w1t + (size_t)e * HID_DIM * IN_DIM;   // [h][i]
    const h16* w2e = w2t + (size_t)e * OUT_DIM * HID_DIM;  // [o][h]
    const float* b1e = b1 + e * HID_DIM;

    v8f oacc[2][8];
#pragma unroll
    for (int mh = 0; mh < 2; ++mh)
#pragma unroll
        for (int j = 0; j < 8; ++j) oacc[mh][j] = (v8f)(0.f);

    for (int hc = 0; hc < HID_DIM / 128; ++hc) {
        const int h0 = hc * 128;

        // ---- GEMM1 (transposed): H^T[128h][32m] = W1c^T @ X^T --------------
        v8f ca = (v8f)(0.f), cb = (v8f)(0.f);
        stage_w1_async(Bs, w1e, h0, 0, tid);
        for (int k = 0; k < IN_DIM / 32; ++k) {
            if (k + 1 < IN_DIM / 32) {
                stage_w1_async(Bs + ((k + 1) & 1) * 128 * BS_STRIDE, w1e, h0, k + 1, tid);
                WAIT_ASYNC(2);             // drain stage(k) (+X on k==0), keep stage(k+1)
            } else {
                WAIT_ASYNC(0);
            }
            __syncthreads();
            const h16* buf = Bs + (k & 1) * 128 * BS_STRIDE;
            const int k0 = k * 32;
            v16h aw  = ld_a(buf + (16 * wv + ln16) * BS_STRIDE + 16 * lh);
            v16h bx0 = ld_b(Xs + ln16 * XS_STRIDE + k0 + 16 * lh);
            v16h bx1 = ld_b(Xs + (16 + ln16) * XS_STRIDE + k0 + 16 * lh);
            ca = wmma16x16x32(aw, bx0, ca);
            cb = wmma16x16x32(aw, bx1, cb);
        }
        // C' lane: rows h = 16*wv + 8*lh + r (contiguous!), col m = 16*mh+ln16
        {
            const int hl = 16 * wv + 8 * lh;
            float bb[8];
            v4f blo = *(const v4f*)(b1e + h0 + hl);
            v4f bhi = *(const v4f*)(b1e + h0 + hl + 4);
#pragma unroll
            for (int q = 0; q < 4; ++q) { bb[q] = blo[q]; bb[4 + q] = bhi[q]; }
            *(v8h*)(Hs + ln16 * HS_STRIDE + hl)        = relu_pack(ca, bb);
            *(v8h*)(Hs + (16 + ln16) * HS_STRIDE + hl) = relu_pack(cb, bb);
        }

        // ---- GEMM2: out[32][1024] += Hchunk @ W2[h0:h0+128, :] -------------
        stage_w2_async(W2s, w2e, h0, 0, tid);
        for (int k2 = 0; k2 < 4; ++k2) {
            if (k2 + 1 < 4) {
                stage_w2_async(W2s + ((k2 + 1) & 1) * 1024 * BS_STRIDE, w2e, h0, k2 + 1, tid);
                WAIT_ASYNC(16);
            } else {
                WAIT_ASYNC(0);
            }
            __syncthreads();               // also makes Hs writes visible (k2==0)
            const h16* buf = W2s + (k2 & 1) * 1024 * BS_STRIDE;
            const int koff = k2 * 32 + 16 * lh;
            v16h a0 = ld_a(Hs + ln16 * HS_STRIDE + koff);
            v16h a1 = ld_a(Hs + (16 + ln16) * HS_STRIDE + koff);
#pragma unroll
            for (int j = 0; j < 8; ++j) {
                v16h b = ld_b(buf + (128 * wv + 16 * j + ln16) * BS_STRIDE + 16 * lh);
                oacc[0][j] = wmma16x16x32(a0, b, oacc[0][j]);
                oacc[1][j] = wmma16x16x32(a1, b, oacc[1][j]);
            }
        }
        __syncthreads();                   // protect Hs/Bs reuse next chunk
    }

    // ---- epilogue: exactly 2 commutative f32 atomics per out element -------
    const int mb = lh * 8;
#pragma unroll
    for (int j = 0; j < 8; ++j) {
        const int o = 128 * wv + 16 * j + ln16;
        const float b2v = b2[e * OUT_DIM + o];
#pragma unroll
        for (int mh = 0; mh < 2; ++mh)
#pragma unroll
            for (int r = 0; r < 8; ++r) {
                const int m = mh * 16 + mb + r;
                if (row0 + m < nrows) {
                    float val = wt[m] * (oacc[mh][j][r] + b2v);
                    atomicAdd(out + (size_t)tok[m] * OUT_DIM + o, val);
                }
            }
    }
}

// ---------------- K2b: fallback (inline fp32->f16 staging) ------------------
__device__ __forceinline__ void stage_w1_f32(h16* dst, const float* W1e,
                                             int h0, int k, int tid) {
    const int kk = tid >> 3;
    const int c0 = (tid & 7) * 16;
    const float* src = W1e + (size_t)(k * 32 + kk) * HID_DIM + h0 + c0;
#pragma unroll
    for (int j = 0; j < 4; ++j) {
        v4f v = *(const v4f*)(src + j * 4);
#pragma unroll
        for (int q = 0; q < 4; ++q)
            dst[(c0 + j * 4 + q) * BS_STRIDE + kk] = (h16)v[q];
    }
}
__device__ __forceinline__ void stage_w2_f32(h16* dst, const float* W2e,
                                             int h0, int k2, int tid) {
    const int kk = tid >> 3;
    const int g  = tid & 7;
    const float* src = W2e + (size_t)(h0 + k2 * 32 + kk) * OUT_DIM;
#pragma unroll
    for (int j = 0; j < 32; ++j) {
        int o0 = j * 32 + g * 4;
        v4f v = *(const v4f*)(src + o0);
#pragma unroll
        for (int q = 0; q < 4; ++q)
            dst[(o0 + q) * BS_STRIDE + kk] = (h16)v[q];
    }
}

__global__ __launch_bounds__(256) void k_moe_direct(
        const float* __restrict__ x,
        const float* __restrict__ W1, const float* __restrict__ b1,
        const float* __restrict__ W2, const float* __restrict__ b2,
        const int* __restrict__ counts, const int* __restrict__ tlist,
        const float* __restrict__ wlist, float* __restrict__ out) {
    const int e     = blockIdx.x >> 8;
    const int tile  = blockIdx.x & 255;
    const int nrows = counts[e];
    const int row0  = tile * MTILE;
    if (row0 >= nrows) return;

    extern __shared__ h16 smem[];
    h16* Xs  = smem;
    h16* Hs  = Xs + XS_HALFS;
    h16* Bs  = Hs + HS_HALFS;
    h16* W2s = Bs + BS_HALFS;
    int*   tok = (int*)(W2s + W2S_HALFS);
    float* wt  = (float*)(tok + 32);

    const int tid  = threadIdx.x;
    const int wv   = tid >> 5;
    const int lane = tid & 31;
    const int ln16 = lane & 15;
    const int lh   = lane >> 4;

    if (tid < 32) {
        int r = row0 + tid;
        int tk = 0; float w = 0.f;
        if (r < nrows) { tk = tlist[e * NTOK + r]; w = wlist[e * NTOK + r]; }
        tok[tid] = tk; wt[tid] = w;
    }
    __syncthreads();
    {
        const int m = tid >> 3, g = tid & 7;
        const float* xr = x + (size_t)tok[m] * IN_DIM;
        h16* dst = Xs + m * XS_STRIDE;
#pragma unroll 4
        for (int j = 0; j < 32; ++j) {
            int c4 = g + 8 * j;
            v4f v = *(const v4f*)(xr + c4 * 4);
            v4h h; h[0] = (h16)v[0]; h[1] = (h16)v[1];
            h[2] = (h16)v[2]; h[3] = (h16)v[3];
            *(v4h*)(dst + c4 * 4) = h;
        }
    }

    const float* W1e = W1 + (size_t)e * IN_DIM * HID_DIM;
    const float* W2e = W2 + (size_t)e * HID_DIM * OUT_DIM;

    v8f oacc[2][8];
#pragma unroll
    for (int mh = 0; mh < 2; ++mh)
#pragma unroll
        for (int j = 0; j < 8; ++j) oacc[mh][j] = (v8f)(0.f);

    for (int hc = 0; hc < HID_DIM / 128; ++hc) {
        const int h0 = hc * 128;
        v8f h0acc = (v8f)(0.f), h1acc = (v8f)(0.f);
        stage_w1_f32(Bs, W1e, h0, 0, tid);
        for (int k = 0; k < IN_DIM / 32; ++k) {
            __syncthreads();
            const h16* buf = Bs + (k & 1) * 128 * BS_STRIDE;
            const int koff = k * 32 + 16 * lh;
            v16h a0 = ld_a(Xs + ln16 * XS_STRIDE + koff);
            v16h a1 = ld_a(Xs + (ln16 + 16) * XS_STRIDE + koff);
            v16h b  = ld_b(buf + (16 * wv + ln16) * BS_STRIDE + 16 * lh);
            if (k + 1 < IN_DIM / 32)
                stage_w1_f32(Bs + ((k + 1) & 1) * 128 * BS_STRIDE, W1e, h0, k + 1, tid);
            h0acc = wmma16x16x32(a0, b, h0acc);
            h1acc = wmma16x16x32(a1, b, h1acc);
        }
        __syncthreads();
        {
            const int ncl = 16 * wv + ln16;
            const float bv = b1[e * HID_DIM + h0 + ncl];
            const int mb = lh * 8;
#pragma unroll
            for (int r = 0; r < 8; ++r) {
                Hs[(mb + r) * HS_STRIDE + ncl]      = (h16)fmaxf(h0acc[r] + bv, 0.f);
                Hs[(16 + mb + r) * HS_STRIDE + ncl] = (h16)fmaxf(h1acc[r] + bv, 0.f);
            }
        }
        stage_w2_f32(W2s, W2e, h0, 0, tid);
        for (int k2 = 0; k2 < 4; ++k2) {
            __syncthreads();
            const h16* buf = W2s + (k2 & 1) * 1024 * BS_STRIDE;
            const int koff = k2 * 32 + 16 * lh;
            v16h a0 = ld_a(Hs + ln16 * HS_STRIDE + koff);
            v16h a1 = ld_a(Hs + (ln16 + 16) * HS_STRIDE + koff);
            if (k2 + 1 < 4)
                stage_w2_f32(W2s + ((k2 + 1) & 1) * 1024 * BS_STRIDE, W2e, h0, k2 + 1, tid);
#pragma unroll
            for (int j = 0; j < 8; ++j) {
                v16h b = ld_b(buf + (128 * wv + 16 * j + ln16) * BS_STRIDE + 16 * lh);
                oacc[0][j] = wmma16x16x32(a0, b, oacc[0][j]);
                oacc[1][j] = wmma16x16x32(a1, b, oacc[1][j]);
            }
        }
        __syncthreads();
    }

    const int mb = lh * 8;
#pragma unroll
    for (int j = 0; j < 8; ++j) {
        const int o = 128 * wv + 16 * j + ln16;
        const float b2v = b2[e * OUT_DIM + o];
#pragma unroll
        for (int mh = 0; mh < 2; ++mh)
#pragma unroll
            for (int r = 0; r < 8; ++r) {
                const int m = mh * 16 + mb + r;
                if (row0 + m < nrows) {
                    float val = wt[m] * (oacc[mh][j][r] + b2v);
                    atomicAdd(out + (size_t)tok[m] * OUT_DIM + o, val);
                }
            }
    }
}

// ---------------------------------------------------------------------------
extern "C" void kernel_launch(void* const* d_in, const int* in_sizes, int n_in,
                              void* d_out, int out_size, void* d_ws, size_t ws_size,
                              hipStream_t stream) {
    const float* x  = (const float*)d_in[0];
    const float* Wg = (const float*)d_in[1];
    const float* bg = (const float*)d_in[2];
    const float* W1 = (const float*)d_in[3];
    const float* b1 = (const float*)d_in[4];
    const float* W2 = (const float*)d_in[5];
    const float* b2 = (const float*)d_in[6];
    float* out = (float*)d_out;

    char* ws = (char*)d_ws;
    int*   counts = (int*)ws;
    int*   tlist  = (int*)(ws + 64);
    float* wlist  = (float*)(ws + 64 + (size_t)NEXP * NTOK * 4);

    const size_t ROUTE_OFF = (size_t)512 << 10;
    h16* xh  = (h16*)(ws + ROUTE_OFF);
    h16* w1t = xh + (size_t)NTOK * IN_DIM;
    h16* w2t = w1t + (size_t)NEXP * IN_DIM * HID_DIM;
    const size_t need = ROUTE_OFF +
        2 * ((size_t)NTOK * IN_DIM + 2 * (size_t)NEXP * IN_DIM * HID_DIM);

    (void)in_sizes; (void)n_in; (void)out_size;

    (void)hipFuncSetAttribute((const void*)k_moe_async,
                              hipFuncAttributeMaxDynamicSharedMemorySize, SMEM_BYTES);
    (void)hipFuncSetAttribute((const void*)k_moe_direct,
                              hipFuncAttributeMaxDynamicSharedMemorySize, SMEM_BYTES);

    k_zero<<<8192, 256, 0, stream>>>(out, counts);
    k_gate<<<NTOK / 8, 256, 0, stream>>>(x, Wg, bg, counts, tlist, wlist);

    if (ws_size >= need) {
        k_cvt_x<<<(NTOK * IN_DIM / 4) / 256, 256, 0, stream>>>(x, xh);
        k_cvt_t<<<NEXP * (IN_DIM / 32) * (HID_DIM / 32), 256, 0, stream>>>(
            W1, w1t, IN_DIM, HID_DIM);
        k_cvt_t<<<NEXP * (HID_DIM / 32) * (OUT_DIM / 32), 256, 0, stream>>>(
            W2, w2t, HID_DIM, OUT_DIM);
        k_moe_async<<<NEXP * (NTOK / MTILE), 256, SMEM_BYTES, stream>>>(
            xh, w1t, w2t, b1, b2, counts, tlist, wlist, out);
    } else {
        k_moe_direct<<<NEXP * (NTOK / MTILE), 256, SMEM_BYTES, stream>>>(
            x, W1, b1, W2, b2, counts, tlist, wlist, out);
    }
}
